// Truncated_Hausdorff_Distance_35416300323049
// MI455X (gfx1250) — compile-verified, matrix operand order checked by reading, not run on hardware
//
#include <hip/hip_runtime.h>
#include <hip/hip_bf16.h>
#include <hip/hip_fp16.h>

// Problem constants (from reference)
#define BB   8
#define CCL  4
#define HH   96
#define WWD  96
#define HWP  (HH * WWD)   // 9216

typedef __attribute__((ext_vector_type(16))) _Float16 v16h;
typedef __attribute__((ext_vector_type(8)))  float    v8f;

#define INF_BITS 0x7F800000u

static __device__ __forceinline__ unsigned umin32(unsigned a, unsigned b) {
    return a < b ? a : b;   // -> v_min_u32 (no float canonicalization)
}

// ---------------------------------------------------------------------------
// Kernel A: pred_cls = argmax over C of logits (first-max tie break, like jnp)
// ---------------------------------------------------------------------------
__global__ void thd_argmax_kernel(const float* __restrict__ pred,
                                  int* __restrict__ cls) {
    int idx = blockIdx.x * blockDim.x + threadIdx.x;   // over B*HW
    if (idx >= BB * HWP) return;
    int b = idx / HWP, p = idx % HWP;
    const float* base = pred + (size_t)b * CCL * HWP + p;
    float best = base[0];
    int bc = 0;
#pragma unroll
    for (int c = 1; c < CCL; ++c) {
        float v = base[(size_t)c * HWP];
        if (v > best) { best = v; bc = c; }
    }
    cls[idx] = bc;
}

// ---------------------------------------------------------------------------
// Kernel B: per-class nearest-distance fields via WMMA-fused pair distances.
// fields[src][b][c][i] = min_{j : clsmap_src[b][j]==c} dist(i, j)
//   src 0: clsmap = labels   (-> fwd_min fields)
//   src 1: clsmap = pred_cls (-> rev_min fields)
// One wave per (b, src, 16-pixel i-tile); loops 6 column-tiles x 96 rows.
// A row  = (-y_i, -x_i, 0..)  (negated so no f16 NEG modifier is needed)
// B col  = (2y_j, 2x_j, 0..)
// C tile = sq_i + sq_j  (per-element, fed as WMMA accumulator input)
// => WMMA output IS d2 = sq_i + sq_j - 2*(y_i y_j + x_i x_j), exactly
// (integer coords exact in f16, f32 accum, magnitudes < 2^24). d2 >= 0
// exactly, so its f32 bit pattern is order-isomorphic to unsigned int:
// min-reduce with v_min_u32 (no canonicalize), sentinel/init = +inf bits.
// One precise sqrtf per output in the epilogue (sqrt is monotone:
// sqrt(min d2) == min sqrt(d2) bit-exactly).
// ---------------------------------------------------------------------------
__global__ void __launch_bounds__(256)
thd_distfield_kernel(const int* __restrict__ labels,
                     const int* __restrict__ predcls,
                     float* __restrict__ fields) {
    const int b    = blockIdx.y;
    const int src  = blockIdx.z;
    const int wave = threadIdx.x >> 5;
    const int lane = threadIdx.x & 31;
    const int t    = lane & 15;      // N-index / row pixel within tile
    const int half = lane >> 4;      // lane half (16..31 => second half)
    const int tile = blockIdx.x * 8 + wave;
    const int i0   = tile * 16;

    const int* clsmap = (src == 0 ? labels : predcls) + b * HWP;

    // A operand (16x32 f16): lanes 0-15 hold row M=t with K0=-y, K1=-x; rest 0.
    int pi = i0 + t;
    float yi = (float)(pi / WWD), xi = (float)(pi % WWD);
    v16h a = {};
    a[0] = (half == 0) ? (_Float16)(-yi) : (_Float16)0.f;
    a[1] = (half == 0) ? (_Float16)(-xi) : (_Float16)0.f;

    // Squared norms of the 8 C/D rows this lane owns: row m = v + 8*half
    float sqi[8];
#pragma unroll
    for (int v = 0; v < 8; ++v) {
        int pm = i0 + v + 8 * half;
        float ym = (float)(pm / WWD), xm = (float)(pm % WWD);
        sqi[v] = ym * ym + xm * xm;
    }

    // Min accumulators: f32 bit patterns as unsigned (init = +inf bits).
    unsigned acc[CCL][8];
#pragma unroll
    for (int c = 0; c < CCL; ++c)
#pragma unroll
        for (int v = 0; v < 8; ++v) acc[c][v] = INF_BITS;

    // Single B operand tuple; 7 of its 8 VGPRs are loop-invariant zeros.
    v16h bv = {};

    // Outer: 6 distinct column-x patterns; inner: 96 rows.
    for (int jc = 0; jc < 6; ++jc) {
        float xq = (float)(jc * 16 + t);
        _Float16 bxh = (half == 0) ? (_Float16)(2.f * xq) : (_Float16)0.f;
        float sqx = xq * xq;
        bv[1] = bxh;

        const int* colcls = clsmap + jc * 16 + t;
        for (int jy = 0; jy < HH; ++jy) {
            float fy = (float)jy;                 // scalar-uniform per row
            bv[0] = (half == 0) ? (_Float16)(2.f * fy) : (_Float16)0.f;
            float sqj = fy * fy + sqx;            // v_fma with scalar fy
            int cj = colcls[jy * WWD];            // class of column pixel (N=t)

            // C input = sq_i + sq_j per element; WMMA output = d2.
            v8f cacc;
#pragma unroll
            for (int v = 0; v < 8; ++v) cacc[v] = sqi[v] + sqj;

            cacc = __builtin_amdgcn_wmma_f32_16x16x32_f16(
                /*neg_a=*/false, a, /*neg_b=*/false, bv,
                /*c_mod=*/(short)0, cacc, /*reuse_a=*/false, /*reuse_b=*/false);

#pragma unroll
            for (int v = 0; v < 8; ++v) {
                unsigned u = __float_as_uint(cacc[v]);   // d2, exact, >= 0
#pragma unroll
                for (int c = 0; c < CCL; ++c)
                    acc[c][v] = umin32(acc[c][v], (cj == c) ? u : INF_BITS);
            }
        }
    }

    // Min across the 16 lanes of each half (same M row, different N columns)
#pragma unroll
    for (int c = 0; c < CCL; ++c)
#pragma unroll
        for (int v = 0; v < 8; ++v) {
            unsigned x = acc[c][v];
#pragma unroll
            for (int m = 1; m < 16; m <<= 1)
                x = umin32(x, (unsigned)__shfl_xor((int)x, m, 32));
            acc[c][v] = x;
        }

    if (t == 0) {
#pragma unroll
        for (int c = 0; c < CCL; ++c)
#pragma unroll
            for (int v = 0; v < 8; ++v)
                fields[(((size_t)src * BB + b) * CCL + c) * HWP + i0 + v + 8 * half] =
                    sqrtf(fmaxf(__uint_as_float(acc[c][v]), 0.f));
    }
}

// ---------------------------------------------------------------------------
// Kernel C: masked 95th percentile (linear interpolation) per (b, c, dir)
// via LDS radix-select on float bit patterns (non-negative floats are
// order-preserving as uints; unmasked -> 0xFFFFFFFF sentinel above +inf).
// ---------------------------------------------------------------------------
__device__ unsigned thd_radix_select(const unsigned* keys, int k,
                                     int tid, int nth, int* s_cnt) {
    unsigned cand = 0;
    for (int bit = 31; bit >= 0; --bit) {
        unsigned hiMask = (bit == 31) ? 0u : (0xFFFFFFFFu << (bit + 1));
        __syncthreads();
        if (tid == 0) *s_cnt = 0;
        __syncthreads();
        int local = 0;
        for (int p = tid; p < HWP; p += nth) {
            unsigned v = keys[p];
            local += ((((v ^ cand) & hiMask) == 0u) && (((v >> bit) & 1u) == 0u)) ? 1 : 0;
        }
        atomicAdd(s_cnt, local);
        __syncthreads();
        int cnt0 = *s_cnt;
        if (k >= cnt0) { cand |= (1u << bit); k -= cnt0; }
    }
    return cand;
}

__global__ void __launch_bounds__(256)
thd_percentile_kernel(const int* __restrict__ labels,
                      const int* __restrict__ predcls,
                      const float* __restrict__ fields,
                      float* __restrict__ res) {
    __shared__ unsigned keys[HWP];   // 36 KB
    __shared__ int s_cnt;

    const int tid = threadIdx.x, nth = blockDim.x;
    const int idx  = blockIdx.x;          // 0..63
    const int dir  = idx & 1;             // 0: fwd, 1: rev
    const int pair = idx >> 1;
    const int b = pair / CCL, c = pair % CCL;

    // fwd: values = dist-to-label-set, mask = pred set
    // rev: values = dist-to-pred-set,  mask = label set
    const float* vals = fields + (((size_t)dir * BB + b) * CCL + c) * HWP;
    const int*   msk  = (dir == 0 ? predcls : labels) + b * HWP;

    if (tid == 0) s_cnt = 0;
    __syncthreads();
    int ln = 0;
    for (int p = tid; p < HWP; p += nth) {
        bool inset = (msk[p] == c);
        keys[p] = inset ? __float_as_uint(vals[p]) : 0xFFFFFFFFu;
        ln += inset ? 1 : 0;
    }
    atomicAdd(&s_cnt, ln);
    __syncthreads();
    int n = s_cnt;
    __syncthreads();

    float pos  = 0.95f * (float)(n > 0 ? (n - 1) : 0);
    int   lo   = (int)floorf(pos);
    int   hi   = (int)ceilf(pos);
    float frac = pos - (float)lo;

    unsigned vlo = thd_radix_select(keys, lo, tid, nth, &s_cnt);
    unsigned vhi = (hi == lo) ? vlo : thd_radix_select(keys, hi, tid, nth, &s_cnt);

    if (tid == 0)
        res[(b * CCL + c) * 2 + dir] =
            __uint_as_float(vlo) * (1.f - frac) + __uint_as_float(vhi) * frac;
}

// ---------------------------------------------------------------------------
// Kernel D: batch mean with class-0 zeroing + finalize -> 18 outputs
// out = [MHD(6), FHD(6), RHD(6)], each = [per-class(4), mean, mean(excl c0)]
// ---------------------------------------------------------------------------
__global__ void thd_finalize_kernel(const float* __restrict__ res,
                                    float* __restrict__ out) {
    if (threadIdx.x != 0 || blockIdx.x != 0) return;
    float mf[CCL], mr[CCL], mm[CCL];
    for (int c = 0; c < CCL; ++c) {
        float sf = 0.f, sr = 0.f, sm = 0.f;
        for (int b = 0; b < BB; ++b) {
            float f = res[(b * CCL + c) * 2 + 0];
            float r = res[(b * CCL + c) * 2 + 1];
            float m = fmaxf(f, r);
            if (c != 0) { sf += f; sr += r; sm += m; }   // ignore-class zeroed
        }
        mf[c] = sf / (float)BB;
        mr[c] = sr / (float)BB;
        mm[c] = sm / (float)BB;
    }
    const float* arrs[3] = {mm, mf, mr};   // MHD, FHD, RHD
    for (int k = 0; k < 3; ++k) {
        const float* v = arrs[k];
        out[k * 6 + 0] = v[0];
        out[k * 6 + 1] = v[1];
        out[k * 6 + 2] = v[2];
        out[k * 6 + 3] = v[3];
        out[k * 6 + 4] = (v[0] + v[1] + v[2] + v[3]) * 0.25f;
        out[k * 6 + 5] = (v[1] + v[2] + v[3]) * (1.f / 3.f);
    }
}

// ---------------------------------------------------------------------------
extern "C" void kernel_launch(void* const* d_in, const int* in_sizes, int n_in,
                              void* d_out, int out_size, void* d_ws, size_t ws_size,
                              hipStream_t stream) {
    (void)in_sizes; (void)n_in; (void)out_size; (void)ws_size;

    const float* pred   = (const float*)d_in[0];  // [B,C,H,W] f32
    const int*   labels = (const int*)d_in[1];    // [B,H,W] i32

    char* ws = (char*)d_ws;
    int*   predcls = (int*)ws;                                       // B*HW ints
    float* fields  = (float*)(ws + (size_t)BB * HWP * sizeof(int));  // 2*B*C*HW f32
    float* res     = (float*)(ws + (size_t)BB * HWP * sizeof(int)
                                 + (size_t)2 * BB * CCL * HWP * sizeof(float)); // B*C*2

    thd_argmax_kernel<<<(BB * HWP + 255) / 256, 256, 0, stream>>>(pred, predcls);

    dim3 gB(HWP / 16 / 8, BB, 2);   // 72 x 8 x 2 blocks, 8 waves each
    thd_distfield_kernel<<<gB, 256, 0, stream>>>(labels, predcls, fields);

    thd_percentile_kernel<<<BB * CCL * 2, 256, 0, stream>>>(labels, predcls, fields, res);

    thd_finalize_kernel<<<1, 32, 0, stream>>>(res, (float*)d_out);
}
